// LstmModel_25486335935219
// MI455X (gfx1250) — matrix operand order repaired and checked by reading, not compiled, not run on hardware
//
#include <hip/hip_runtime.h>
#include <hip/hip_bf16.h>

// ---------------------------------------------------------------------------
// Types for CDNA5 WMMA (wave32): D(16x16 f32) = A(16x32 bf16) x B(32x16 bf16) + C
// ---------------------------------------------------------------------------
typedef __bf16 bf16;
typedef bf16  v16bf __attribute__((ext_vector_type(16)));
typedef bf16  v8bf  __attribute__((ext_vector_type(8)));
typedef float v8f   __attribute__((ext_vector_type(8)));

#define B_  1024
#define T_  28
#define H_  1024
#define G4_ 4096      // 4*H
#define M_  (B_*T_)   // 28672
#define LDSPAD 40     // padded row stride (halves) for 64-bank conflict avoidance

__device__ __forceinline__ v8f wmma_bf16(v16bf a, v16bf b, v8f c) {
  return __builtin_amdgcn_wmma_f32_16x16x32_bf16(false, a, false, b,
                                                 (short)0, c, false, false);
}

// A fragment from global: 16x32 tile, row m = lane&15, contiguous K order:
//   lane<16 : e=0..15 -> K=k0+e
//   lane>=16: e=0..15 -> K=k0+16+e
__device__ __forceinline__ v16bf load_a(const bf16* A, long rowStride, int k0, int lane) {
  int m = lane & 15, hi = lane >> 4;
  const bf16* p = A + (long)m * rowStride + k0 + hi * 16;
  v8bf lo = *(const v8bf*)p;
  v8bf hh = *(const v8bf*)(p + 8);
  v16bf r;
#pragma unroll
  for (int i = 0; i < 8; ++i) { r[i] = lo[i]; r[i + 8] = hh[i]; }
  return r;
}

// B fragment from an LDS-staged strip laid out as [64 rows][LDSPAD halves],
// row r holding K-halves 0..31 of W row (rowbase + r).  In the contiguous k order shared with load_a the
// lane needs 16 contiguous K-halves starting at 16*(lane>>4) of row (lane&15).
__device__ __forceinline__ v16bf load_b_lds(const bf16* Lb, int row16base, int lane) {
  const bf16* p = Lb + (row16base + (lane & 15)) * LDSPAD + ((lane >> 4) * 16);
  v8bf lo = *(const v8bf*)p;
  v8bf hh = *(const v8bf*)(p + 8);
  v16bf r;
#pragma unroll
  for (int i = 0; i < 8; ++i) { r[i] = lo[i]; r[i + 8] = hh[i]; }
  return r;
}

// Cooperative stage of one 64-row x 32-half W strip into LDS (128 threads:
// thread i copies 16 halves = 32B of row i>>1, half-chunk i&1).
__device__ __forceinline__ void stage_strip(const bf16* __restrict__ wrow_k,
                                            bf16* __restrict__ dst) {
  v8bf a0 = *(const v8bf*)wrow_k;
  v8bf a1 = *(const v8bf*)(wrow_k + 8);
  *(v8bf*)dst       = a0;
  *(v8bf*)(dst + 8) = a1;
}

__device__ __forceinline__ float sigmoidf_(float x) {
  return 1.0f / (1.0f + __expf(-x));
}

// ---------------------------------------------------------------------------
// Utility kernels
// ---------------------------------------------------------------------------
__global__ void fzero(float* __restrict__ p, long n) {
  long i = (long)blockIdx.x * blockDim.x + threadIdx.x;
  if (i < n) p[i] = 0.0f;
}

__global__ void cvt_bf16(const float* __restrict__ in, bf16* __restrict__ out, long n) {
  long i = (long)blockIdx.x * blockDim.x + threadIdx.x;
  if (i < n) out[i] = (bf16)in[i];
}

// Encoder layer 0 input projection: K = IN = 4, scalar fp32 (memory bound).
__global__ void xg_small(const float* __restrict__ x, const float* __restrict__ Wih0,
                         const float* __restrict__ b0, float* __restrict__ xg) {
  long idx = (long)blockIdx.x * blockDim.x + threadIdx.x;
  if (idx >= (long)M_ * G4_) return;
  int  n = (int)(idx & (G4_ - 1));
  long m = idx >> 12;
  const float* xr = x + m * 4;
  const float* wr = Wih0 + (long)n * 4;
  xg[idx] = b0[n] + xr[0] * wr[0] + xr[1] * wr[1] + xr[2] * wr[2] + xr[3] * wr[3];
}

// ---------------------------------------------------------------------------
// xg = Xbf16[M,1024] @ W[4096,1024]^T + bias
// Block = 128 threads (4 waves).  All 4 waves share ONE 64-column strip of W
// (rows n0..n0+63), double-buffered in LDS; each wave owns a different 16-row
// m-tile and keeps 4 f32 accumulators (16x64 per wave, 64x64 per block).
// ---------------------------------------------------------------------------
__global__ void gemm_ih(const bf16* __restrict__ X, const bf16* __restrict__ W,
                        const float* __restrict__ bias, float* __restrict__ xg) {
  __shared__ bf16 lsb[2][64 * LDSPAD];
  int lane = threadIdx.x & 31;
  int wv   = threadIdx.x >> 5;                    // 0..3
  int m0   = (blockIdx.x * 4 + wv) * 16;          // 448 m-groups
  int n0   = blockIdx.y * 64;                     // 64 n-strips
  // staging assignment
  int srow  = threadIdx.x >> 1;                   // 0..63
  int shalf = threadIdx.x & 1;                    // 0..1
  const bf16* wsrc = W + (long)(n0 + srow) * H_ + shalf * 16;
  bf16* sdst0 = &lsb[0][srow * LDSPAD + shalf * 16];
  bf16* sdst1 = &lsb[1][srow * LDSPAD + shalf * 16];

  v8f acc[4] = {};
  const bf16* Abase = X + (long)m0 * H_;

  stage_strip(wsrc, sdst0);
  __syncthreads();
#pragma unroll 4
  for (int it = 0; it < 32; ++it) {
    int k0 = it * 32;
    if (it + 1 < 32) stage_strip(wsrc + k0 + 32, (it & 1) ? sdst0 : sdst1);
    const bf16* Lb = lsb[it & 1];
    v16bf a = load_a(Abase, H_, k0, lane);
    __builtin_prefetch(Abase + (long)(lane & 15) * H_ + k0 + 32, 0, 0);
#pragma unroll
    for (int j = 0; j < 4; ++j) {
      v16bf b = load_b_lds(Lb, 16 * j, lane);
      acc[j] = wmma_bf16(a, b, acc[j]);
    }
    __syncthreads();
  }

  int col = lane & 15, hi = lane >> 4;
#pragma unroll
  for (int j = 0; j < 4; ++j) {
    int n = n0 + 16 * j + col;
    float bv = bias[n];
#pragma unroll
    for (int v = 0; v < 8; ++v) {
      int m = m0 + v + 8 * hi;
      xg[(long)m * G4_ + n] = acc[j][v] + bv;
    }
  }
}

// ---------------------------------------------------------------------------
// One recurrence step: g = xg[:,t,:] + h_{t-1} @ Whh^T ; gates ; c,h update.
// Block = 4 waves sharing ONE hidden 16-column strip: the 64 Whh rows
// {g*H + n0 + n} for the 4 gates are staged in LDS (double-buffered); each
// wave owns a different 16-batch-row m-tile.  hbuf/out alias the [B,T,H]
// bf16 buffer (read slice t-1, write slice t).
// ---------------------------------------------------------------------------
__global__ void lstm_step(const float* __restrict__ xg, const bf16* hbuf,
                          const bf16* __restrict__ Whh, float* __restrict__ c,
                          bf16* out, int t, int first) {
  __shared__ bf16 lsb[2][64 * LDSPAD];
  int lane = threadIdx.x & 31;
  int wv   = threadIdx.x >> 5;
  int n0   = (blockIdx.x & 63) * 16;                    // hidden tile (shared)
  int mg   = blockIdx.x >> 6;                           // 0..15
  int m0   = (mg * 4 + wv) * 16;                        // batch tile per wave

  v8f acc[4] = {};
  if (!first) {
    int srow  = threadIdx.x >> 1;                       // 0..63 = gate*16 + n
    int shalf = threadIdx.x & 1;
    int sg = srow >> 4, sn = srow & 15;
    const bf16* wsrc = Whh + (long)(sg * H_ + n0 + sn) * H_ + shalf * 16;
    bf16* sdst0 = &lsb[0][srow * LDSPAD + shalf * 16];
    bf16* sdst1 = &lsb[1][srow * LDSPAD + shalf * 16];

    const bf16* Abase = hbuf + ((long)m0 * T_ + (t - 1)) * H_;
    stage_strip(wsrc, sdst0);
    __syncthreads();
#pragma unroll 4
    for (int it = 0; it < 32; ++it) {
      int k0 = it * 32;
      if (it + 1 < 32) stage_strip(wsrc + k0 + 32, (it & 1) ? sdst0 : sdst1);
      const bf16* Lb = lsb[it & 1];
      v16bf a = load_a(Abase, (long)T_ * H_, k0, lane);
      __builtin_prefetch(Abase + (long)(lane & 15) * T_ * H_ + k0 + 32, 0, 0);
#pragma unroll
      for (int g = 0; g < 4; ++g) {
        v16bf b = load_b_lds(Lb, 16 * g, lane);
        acc[g] = wmma_bf16(a, b, acc[g]);
      }
      __syncthreads();
    }
  }

  int col = n0 + (lane & 15), hi = lane >> 4;
#pragma unroll
  for (int v = 0; v < 8; ++v) {
    int  b    = m0 + v + 8 * hi;
    long base = ((long)b * T_ + t) * G4_;
    float gi = xg[base + 0 * H_ + col] + acc[0][v];
    float gf = xg[base + 1 * H_ + col] + acc[1][v];
    float gg = xg[base + 2 * H_ + col] + acc[2][v];
    float go = xg[base + 3 * H_ + col] + acc[3][v];
    long cidx = (long)b * H_ + col;
    float cc = c[cidx];
    cc = sigmoidf_(gf) * cc + sigmoidf_(gi) * tanhf(gg);
    c[cidx] = cc;
    float h = sigmoidf_(go) * tanhf(cc);
    out[((long)b * T_ + t) * H_ + col] = (bf16)h;
  }
}

// ---------------------------------------------------------------------------
// Decoder layer-0 input projection: [M,1024] @ [16,1024]^T + b  (WMMA, N=16)
// ---------------------------------------------------------------------------
__global__ void gemm_dec0(const bf16* __restrict__ X, const bf16* __restrict__ W,
                          const float* __restrict__ bias, float* __restrict__ xg16) {
  int lane = threadIdx.x & 31;
  int wv   = threadIdx.x >> 5;
  int m0   = (blockIdx.x * 4 + wv) * 16;
  v8f acc = {};
  const bf16* Abase = X + (long)m0 * H_;
  for (int k0 = 0; k0 < H_; k0 += 32) {
    v16bf a = load_a(Abase, H_, k0, lane);
    int n = lane & 15, hi = lane >> 4;
    const bf16* p = W + (long)n * H_ + k0 + hi * 16;
    v16bf b = *(const v16bf*)p;
    acc = wmma_bf16(a, b, acc);
  }
  int col = lane & 15, hi = lane >> 4;
  float bv = bias[col];
#pragma unroll
  for (int v = 0; v < 8; ++v) {
    int m = m0 + v + 8 * hi;
    xg16[(long)m * 16 + col] = acc[v] + bv;
  }
}

// Decoder layer-0 recurrence: hidden=4, one thread per batch row.
__global__ void dec_rec0(const float* __restrict__ xg16, const float* __restrict__ Whh,
                         float* __restrict__ out) {
  int b = blockIdx.x * blockDim.x + threadIdx.x;
  if (b >= B_) return;
  float h[4] = {0, 0, 0, 0}, cst[4] = {0, 0, 0, 0};
  float w[64];
#pragma unroll
  for (int i = 0; i < 64; ++i) w[i] = Whh[i];
  for (int t = 0; t < T_; ++t) {
    const float* g0 = xg16 + ((long)b * T_ + t) * 16;
    float g[16];
#pragma unroll
    for (int n = 0; n < 16; ++n) {
      float s = g0[n];
#pragma unroll
      for (int k = 0; k < 4; ++k) s += h[k] * w[n * 4 + k];
      g[n] = s;
    }
#pragma unroll
    for (int j = 0; j < 4; ++j) {
      float cc = sigmoidf_(g[4 + j]) * cst[j] + sigmoidf_(g[j]) * tanhf(g[8 + j]);
      cst[j] = cc;
      h[j] = sigmoidf_(g[12 + j]) * tanhf(cc);
      out[((long)b * T_ + t) * 4 + j] = h[j];
    }
  }
}

// Decoder layers 1..7: IN->IN LSTM fused (ih + recurrence), thread per batch.
__global__ void dec_layer(const float* __restrict__ in, const float* __restrict__ Wih,
                          const float* __restrict__ Whh, const float* __restrict__ bias,
                          float* __restrict__ out) {
  int b = blockIdx.x * blockDim.x + threadIdx.x;
  if (b >= B_) return;
  float wi[64], wh[64], bs[16];
#pragma unroll
  for (int i = 0; i < 64; ++i) { wi[i] = Wih[i]; wh[i] = Whh[i]; }
#pragma unroll
  for (int i = 0; i < 16; ++i) bs[i] = bias[i];
  float h[4] = {0, 0, 0, 0}, cst[4] = {0, 0, 0, 0};
  for (int t = 0; t < T_; ++t) {
    const float* xr = in + ((long)b * T_ + t) * 4;
    float g[16];
#pragma unroll
    for (int n = 0; n < 16; ++n) {
      float s = bs[n];
#pragma unroll
      for (int k = 0; k < 4; ++k) s += xr[k] * wi[n * 4 + k] + h[k] * wh[n * 4 + k];
      g[n] = s;
    }
#pragma unroll
    for (int j = 0; j < 4; ++j) {
      float cc = sigmoidf_(g[4 + j]) * cst[j] + sigmoidf_(g[j]) * tanhf(g[8 + j]);
      cst[j] = cc;
      h[j] = sigmoidf_(g[12 + j]) * tanhf(cc);
      out[((long)b * T_ + t) * 4 + j] = h[j];
    }
  }
}

// Heads: [B,112] @ fc1^T + b1 -> [B,64] @ fc2^T + b2 -> [B,1]
__global__ void head_k(const float* __restrict__ in, const float* __restrict__ w1,
                       const float* __restrict__ b1, const float* __restrict__ w2,
                       const float* __restrict__ b2, float* __restrict__ out) {
  int b = blockIdx.x * blockDim.x + threadIdx.x;
  if (b >= B_) return;
  const float* xr = in + (long)b * 112;
  float acc2 = b2[0];
  for (int i = 0; i < 64; ++i) {
    float s = b1[i];
    const float* wr = w1 + (long)i * 112;
#pragma unroll 4
    for (int k = 0; k < 112; ++k) s += xr[k] * wr[k];
    acc2 += s * w2[i];
  }
  out[b] = acc2;
}

// ---------------------------------------------------------------------------
extern "C" void kernel_launch(void* const* d_in, const int* in_sizes, int n_in,
                              void* d_out, int out_size, void* d_ws, size_t ws_size,
                              hipStream_t stream) {
  (void)in_sizes; (void)n_in; (void)out_size; (void)ws_size;
  const float* x     = (const float*)d_in[0];
  const float* eWih0 = (const float*)d_in[1];
  const float* eWhh0 = (const float*)d_in[2];
  const float* eb0   = (const float*)d_in[3];
  const float* eWih  = (const float*)d_in[4];
  const float* eWhh  = (const float*)d_in[5];
  const float* eb    = (const float*)d_in[6];
  const float* dWih0 = (const float*)d_in[7];
  const float* dWhh0 = (const float*)d_in[8];
  const float* db0   = (const float*)d_in[9];
  const float* dWih  = (const float*)d_in[10];
  const float* dWhh  = (const float*)d_in[11];
  const float* db    = (const float*)d_in[12];
  const float* fc1w  = (const float*)d_in[13];
  const float* fc1b  = (const float*)d_in[14];
  const float* fc2w  = (const float*)d_in[15];
  const float* fc2b  = (const float*)d_in[16];
  float* out = (float*)d_out;

  char* ws = (char*)d_ws;
  size_t off = 0;
  auto alloc = [&](size_t bytes) -> char* {
    char* p = ws + off;
    off = (off + bytes + 255) & ~(size_t)255;
    return p;
  };
  float* xg     = (float*)alloc((size_t)M_ * G4_ * 4);   // 448 MB
  bf16*  act0   = (bf16*) alloc((size_t)M_ * H_ * 2);    // 56 MB
  bf16*  act1   = (bf16*) alloc((size_t)M_ * H_ * 2);    // 56 MB
  float* cbuf   = (float*)alloc((size_t)B_ * H_ * 4);    // 4 MB
  bf16*  wihb   = (bf16*) alloc((size_t)G4_ * H_ * 2);   // 8 MB
  bf16*  whhb   = (bf16*) alloc((size_t)G4_ * H_ * 2);   // 8 MB
  bf16*  dwih0b = (bf16*) alloc((size_t)16 * H_ * 2);
  float* xg16   = (float*)alloc((size_t)M_ * 16 * 4);
  float* dbuf0  = (float*)alloc((size_t)M_ * 4 * 4);
  float* dbuf1  = (float*)alloc((size_t)M_ * 4 * 4);

  const long WN = (long)G4_ * H_;   // weight elements per matrix
  const long CN = (long)B_ * H_;

  // ---------------- Encoder layer 0 (IN=4 -> H) ----------------
  fzero<<<(unsigned)((CN + 255) / 256), 256, 0, stream>>>(cbuf, CN);
  {
    long n = (long)M_ * G4_;
    xg_small<<<(unsigned)((n + 255) / 256), 256, 0, stream>>>(x, eWih0, eb0, xg);
  }
  cvt_bf16<<<(unsigned)((WN + 255) / 256), 256, 0, stream>>>(eWhh0, whhb, WN);
  for (int t = 0; t < T_; ++t)
    lstm_step<<<1024, 128, 0, stream>>>(xg, act0, whhb, cbuf, act0, t, t == 0);

  // ---------------- Encoder layers 1..7 (H -> H) ----------------
  bf16 *cur = act0, *nxt = act1;
  for (int l = 0; l < 7; ++l) {
    fzero<<<(unsigned)((CN + 255) / 256), 256, 0, stream>>>(cbuf, CN);
    cvt_bf16<<<(unsigned)((WN + 255) / 256), 256, 0, stream>>>(eWih + (long)l * WN, wihb, WN);
    cvt_bf16<<<(unsigned)((WN + 255) / 256), 256, 0, stream>>>(eWhh + (long)l * WN, whhb, WN);
    gemm_ih<<<dim3(M_ / 64, G4_ / 64), 128, 0, stream>>>(cur, wihb, eb + (long)l * G4_, xg);
    for (int t = 0; t < T_; ++t)
      lstm_step<<<1024, 128, 0, stream>>>(xg, nxt, whhb, cbuf, nxt, t, t == 0);
    bf16* tmp = cur; cur = nxt; nxt = tmp;
  }

  // ---------------- Decoder ----------------
  cvt_bf16<<<(unsigned)((16L * H_ + 255) / 256), 256, 0, stream>>>(dWih0, dwih0b, 16L * H_);
  gemm_dec0<<<M_ / 16 / 4, 128, 0, stream>>>(cur, dwih0b, db0, xg16);
  dec_rec0<<<B_ / 256, 256, 0, stream>>>(xg16, dWhh0, dbuf0);
  float *dc = dbuf0, *dn = dbuf1;
  for (int l = 0; l < 7; ++l) {
    dec_layer<<<B_ / 256, 256, 0, stream>>>(dc, dWih + (long)l * 64, dWhh + (long)l * 64,
                                            db + (long)l * 16, dn);
    float* tmp = dc; dc = dn; dn = tmp;
  }

  // ---------------- Heads ----------------
  head_k<<<B_ / 256, 256, 0, stream>>>(dc, fc1w, fc1b, fc2w, fc2b, out);
}
